// MAGIEncoder_55052890800725
// MI455X (gfx1250) — matrix operand order repaired and checked by reading, not compiled, not run on hardware
//
#include <hip/hip_runtime.h>
#include <hip/hip_bf16.h>
#include <math.h>

// MAGI GCN encoder for MI455X (gfx1250, wave32, WMMA).
//
// Pipeline: deg -> [GEMM(bf16 WMMA) -> scatter(atomic f32) -> BN+ReLU] x2
//           -> GEMM -> BN+ReLU -> GEMM(+bias) -> row L2 normalize.
// GEMMs use v_wmma_f32_16x16x32_bf16 (f32 accumulate); scatter is the
// bandwidth bottleneck and stays in f32 through L2 atomics.

#define IN_DIM 128
#define HIDC   96
#define PROJC  64
#define BN_EPS 1e-5f
#define L2_EPS 1e-12f

typedef __attribute__((ext_vector_type(16))) __bf16 v16bf;
typedef __attribute__((ext_vector_type(8)))  float  v8f;

static __host__ int cdiv(long long a, long long b) { return (int)((a + b - 1) / b); }

// ---------------- weights: W[K x N] f32 -> Wt[N x K] bf16 (transposed) -----
__global__ void wt_convert(const float* __restrict__ W, __bf16* __restrict__ Wt,
                           int K, int Nw) {
  int t = blockIdx.x * blockDim.x + threadIdx.x;
  if (t >= K * Nw) return;
  int k = t / Nw, n = t % Nw;
  Wt[(size_t)n * K + k] = (__bf16)W[(size_t)k * Nw + n];
}

// ---------------- degrees (self-loop included: init to 1) ------------------
__global__ void deg_init(int* __restrict__ deg, int n) {
  int i = blockIdx.x * blockDim.x + threadIdx.x;
  if (i < n) deg[i] = 1;
}
__global__ void deg_count(const long long* __restrict__ ei, int E, int* __restrict__ deg) {
  int e = blockIdx.x * blockDim.x + threadIdx.x;
  if (e < E) atomicAdd(&deg[(int)ei[(size_t)E + e]], 1);
}
__global__ void deg_rsqrt(const int* __restrict__ deg, float* __restrict__ dis, int n) {
  int i = blockIdx.x * blockDim.x + threadIdx.x;
  if (i < n) dis[i] = rsqrtf((float)deg[i]);
}

// ---------------- GEMM: C[M x Nc] = A[M x Kc] * W[Kc x Nc] (Wt = W^T bf16) -
// One wave computes a 16 x Nc strip of C via Nc/16 WMMA accumulators.
template <int Nc, int Kc>
__global__ void gemm_wmma(const float* __restrict__ A, const __bf16* __restrict__ Wt,
                          const float* __restrict__ bias, float* __restrict__ C,
                          int Mtiles) {
  constexpr int NT = Nc / 16;
  constexpr int KT = Kc / 32;
  const int wave = blockIdx.x * (blockDim.x >> 5) + (threadIdx.x >> 5);
  if (wave >= Mtiles) return;
  const int lane  = threadIdx.x & 31;
  const int half  = lane >> 4;      // 0 or 1
  const int l15   = lane & 15;
  const int arow  = wave * 16 + l15;

  const v8f vzero = {0.f, 0.f, 0.f, 0.f, 0.f, 0.f, 0.f, 0.f};
  v8f acc[NT];
#pragma unroll
  for (int i = 0; i < NT; ++i) acc[i] = vzero;

#pragma unroll
  for (int kt = 0; kt < KT; ++kt) {
    // A fragment (16x32 bf16): lanes 0-15 -> K [0..15], lanes 16-31 -> K [16..31]
    const float* ap = A + (size_t)arow * Kc + kt * 32 + half * 16;
    v16bf a;
#pragma unroll
    for (int i = 0; i < 8; ++i) a[i] = (__bf16)ap[i];
#pragma unroll
    for (int i = 0; i < 8; ++i) a[i + 8] = (__bf16)ap[i + 8];

#pragma unroll
    for (int nt = 0; nt < NT; ++nt) {
      // B fragment (32x16 bf16): lane = column, 16 contiguous K per lane
      const __bf16* bpp = Wt + (size_t)(nt * 16 + l15) * Kc + kt * 32 + half * 16;
      v16bf b = *(const v16bf*)bpp;
      acc[nt] = __builtin_amdgcn_wmma_f32_16x16x32_bf16(
          false, a, false, b, (short)0, acc[nt], false, false);
    }
  }

  // C/D layout: VGPR v, lanes 0-15 -> row v, lanes 16-31 -> row v+8; col = lane&15
  const int r0 = wave * 16 + half * 8;
#pragma unroll
  for (int nt = 0; nt < NT; ++nt) {
    const int col = nt * 16 + l15;
    const float badd = bias ? bias[col] : 0.f;
#pragma unroll
    for (int v = 0; v < 8; ++v)
      C[(size_t)(r0 + v) * Nc + col] = acc[nt][v] + badd;
  }
}

// ---------------- scatter: agg[dst] += xw[src] * dis[src]*dis[dst] ---------
__global__ void scatter_self(const float* __restrict__ xw, const float* __restrict__ dis,
                             float* __restrict__ agg, int n) {
  int t = blockIdx.x * blockDim.x + threadIdx.x;        // n * (HIDC/4)
  if (t >= n * (HIDC / 4)) return;
  int i = t / (HIDC / 4), q = t % (HIDC / 4);
  float w = dis[i] * dis[i];
  float4 v = *((const float4*)(xw + (size_t)i * HIDC) + q);
  float4 o = {v.x * w, v.y * w, v.z * w, v.w * w};
  *((float4*)(agg + (size_t)i * HIDC) + q) = o;
}

__global__ void scatter_edges(const long long* __restrict__ ei, int E,
                              const float* __restrict__ xw, const float* __restrict__ dis,
                              float* __restrict__ agg) {
  int t = blockIdx.x * blockDim.x + threadIdx.x;        // E * (HIDC/4)
  if (t >= E * (HIDC / 4)) return;
  int e = t / (HIDC / 4), q = t % (HIDC / 4);
  int s = (int)ei[e];
  int d = (int)ei[(size_t)E + e];
  float w = dis[s] * dis[d];
  float4 v = *((const float4*)(xw + (size_t)s * HIDC) + q);
  float* dst = agg + (size_t)d * HIDC + q * 4;
  atomicAdd(dst + 0, v.x * w);
  atomicAdd(dst + 1, v.y * w);
  atomicAdd(dst + 2, v.z * w);
  atomicAdd(dst + 3, v.w * w);
}

// ---------------- batch-norm (training stats), bias cancels ----------------
__global__ void stats_init(float* __restrict__ s0, float* __restrict__ s1, int C) {
  int i = threadIdx.x;
  if (i < C) { s0[i] = 0.f; s1[i] = 0.f; }
}
__global__ void bn_stats(const float* __restrict__ h, int n, int C,
                         float* __restrict__ sum, float* __restrict__ sumsq) {
  int c = threadIdx.x;                                   // blockDim == C (96)
  float s = 0.f, ss = 0.f;
  for (int r = blockIdx.x; r < n; r += gridDim.x) {
    float v = h[(size_t)r * C + c];
    s += v; ss += v * v;
  }
  atomicAdd(&sum[c], s);
  atomicAdd(&sumsq[c], ss);
}
__global__ void bn_finalize(const float* __restrict__ sum, const float* __restrict__ sumsq,
                            const float* __restrict__ g, const float* __restrict__ be,
                            float n, int C, float* __restrict__ scale,
                            float* __restrict__ shift) {
  int c = blockIdx.x * blockDim.x + threadIdx.x;
  if (c >= C) return;
  float mean = sum[c] / n;
  float var  = sumsq[c] / n - mean * mean;
  float sc   = g[c] * rsqrtf(var + BN_EPS);
  scale[c] = sc;
  shift[c] = be[c] - sc * mean;
}
__global__ void bn_apply_relu(const float* __restrict__ in, const float* __restrict__ scale,
                              const float* __restrict__ shift, float* __restrict__ out, int n) {
  int t = blockIdx.x * blockDim.x + threadIdx.x;        // n * (HIDC/4)
  if (t >= n * (HIDC / 4)) return;
  int i = t / (HIDC / 4), q = t % (HIDC / 4);
  int c0 = q * 4;
  float4 v = *((const float4*)(in + (size_t)i * HIDC) + q);
  float4 o;
  o.x = fmaxf(0.f, v.x * scale[c0 + 0] + shift[c0 + 0]);
  o.y = fmaxf(0.f, v.y * scale[c0 + 1] + shift[c0 + 1]);
  o.z = fmaxf(0.f, v.z * scale[c0 + 2] + shift[c0 + 2]);
  o.w = fmaxf(0.f, v.w * scale[c0 + 3] + shift[c0 + 3]);
  *((float4*)(out + (size_t)i * HIDC) + q) = o;
}

// ---------------- final row L2 normalize (one wave per row, in place) ------
__global__ void l2norm(float* __restrict__ z, int n) {
  int row = blockIdx.x * (blockDim.x >> 5) + (threadIdx.x >> 5);
  if (row >= n) return;
  int lane = threadIdx.x & 31;
  float* p = z + (size_t)row * PROJC;
  float a = p[lane], b = p[lane + 32];
  float s = a * a + b * b;
#pragma unroll
  for (int m = 16; m >= 1; m >>= 1) s += __shfl_xor(s, m, 32);
  float inv = 1.f / fmaxf(sqrtf(s), L2_EPS);
  p[lane] = a * inv;
  p[lane + 32] = b * inv;
}

// ---------------------------------------------------------------------------
extern "C" void kernel_launch(void* const* d_in, const int* in_sizes, int n_in,
                              void* d_out, int out_size, void* d_ws, size_t ws_size,
                              hipStream_t stream) {
  const float*     x    = (const float*)d_in[0];
  const long long* ei   = (const long long*)d_in[1];
  const float*     W1   = (const float*)d_in[2];
  const float*     g1   = (const float*)d_in[4];
  const float*     be1  = (const float*)d_in[5];
  const float*     W2   = (const float*)d_in[6];
  const float*     g2   = (const float*)d_in[8];
  const float*     be2  = (const float*)d_in[9];
  const float*     Wp1  = (const float*)d_in[10];
  const float*     gp   = (const float*)d_in[12];
  const float*     bep  = (const float*)d_in[13];
  const float*     Wp2  = (const float*)d_in[14];
  const float*     bp2  = (const float*)d_in[15];

  const int N = in_sizes[0] / IN_DIM;   // 50000 (divisible by 16)
  const int E = in_sizes[1] / 2;        // 800000

  // workspace carve-out (256B aligned slices)
  char* ws = (char*)d_ws;
  size_t off = 0;
  auto carve = [&](size_t bytes) -> void* {
    void* p = ws + off;
    off = (off + bytes + 255) & ~(size_t)255;
    return p;
  };
  float*  bufA  = (float*)carve((size_t)N * HIDC * sizeof(float));
  float*  bufB  = (float*)carve((size_t)N * HIDC * sizeof(float));
  float*  dis   = (float*)carve((size_t)N * sizeof(float));
  int*    deg   = (int*)carve((size_t)N * sizeof(int));
  __bf16* Wt1   = (__bf16*)carve((size_t)IN_DIM * HIDC * 2);
  __bf16* Wt2   = (__bf16*)carve((size_t)HIDC * HIDC * 2);
  __bf16* Wtp1  = (__bf16*)carve((size_t)HIDC * HIDC * 2);
  __bf16* Wtp2  = (__bf16*)carve((size_t)HIDC * PROJC * 2);
  float*  ssum  = (float*)carve(HIDC * sizeof(float));
  float*  ssq   = (float*)carve(HIDC * sizeof(float));
  float*  scale = (float*)carve(HIDC * sizeof(float));
  float*  shift = (float*)carve(HIDC * sizeof(float));

  const int T = 256;
  const int Mtiles  = N / 16;
  const int gBlocks = cdiv(Mtiles, 8);            // 8 waves per 256-thread block
  const int rowVec  = cdiv((long long)N * (HIDC / 4), T);
  const int edgVec  = cdiv((long long)E * (HIDC / 4), T);
  float* out = (float*)d_out;

  // ---- weights to bf16 transposed
  wt_convert<<<cdiv(IN_DIM * HIDC, T), T, 0, stream>>>(W1, Wt1, IN_DIM, HIDC);
  wt_convert<<<cdiv(HIDC * HIDC, T), T, 0, stream>>>(W2, Wt2, HIDC, HIDC);
  wt_convert<<<cdiv(HIDC * HIDC, T), T, 0, stream>>>(Wp1, Wtp1, HIDC, HIDC);
  wt_convert<<<cdiv(HIDC * PROJC, T), T, 0, stream>>>(Wp2, Wtp2, HIDC, PROJC);

  // ---- symmetric-norm degrees (self-loops included)
  deg_init<<<cdiv(N, T), T, 0, stream>>>(deg, N);
  deg_count<<<cdiv(E, T), T, 0, stream>>>(ei, E, deg);
  deg_rsqrt<<<cdiv(N, T), T, 0, stream>>>(deg, dis, N);

  // ---- GCN layer 1: x @ W1 -> scatter -> BN+ReLU
  gemm_wmma<HIDC, IN_DIM><<<gBlocks, T, 0, stream>>>(x, Wt1, nullptr, bufA, Mtiles);
  scatter_self<<<rowVec, T, 0, stream>>>(bufA, dis, bufB, N);
  scatter_edges<<<edgVec, T, 0, stream>>>(ei, E, bufA, dis, bufB);
  stats_init<<<1, T, 0, stream>>>(ssum, ssq, HIDC);
  bn_stats<<<512, HIDC, 0, stream>>>(bufB, N, HIDC, ssum, ssq);
  bn_finalize<<<1, 128, 0, stream>>>(ssum, ssq, g1, be1, (float)N, HIDC, scale, shift);
  bn_apply_relu<<<rowVec, T, 0, stream>>>(bufB, scale, shift, bufA, N);

  // ---- GCN layer 2
  gemm_wmma<HIDC, HIDC><<<gBlocks, T, 0, stream>>>(bufA, Wt2, nullptr, bufB, Mtiles);
  scatter_self<<<rowVec, T, 0, stream>>>(bufB, dis, bufA, N);
  scatter_edges<<<edgVec, T, 0, stream>>>(ei, E, bufB, dis, bufA);
  stats_init<<<1, T, 0, stream>>>(ssum, ssq, HIDC);
  bn_stats<<<512, HIDC, 0, stream>>>(bufA, N, HIDC, ssum, ssq);
  bn_finalize<<<1, 128, 0, stream>>>(ssum, ssq, g2, be2, (float)N, HIDC, scale, shift);
  bn_apply_relu<<<rowVec, T, 0, stream>>>(bufA, scale, shift, bufB, N);

  // ---- projector: Linear -> BN -> ReLU -> Linear(+bias) -> L2 normalize
  gemm_wmma<HIDC, HIDC><<<gBlocks, T, 0, stream>>>(bufB, Wtp1, nullptr, bufA, Mtiles);
  stats_init<<<1, T, 0, stream>>>(ssum, ssq, HIDC);
  bn_stats<<<512, HIDC, 0, stream>>>(bufA, N, HIDC, ssum, ssq);
  bn_finalize<<<1, 128, 0, stream>>>(ssum, ssq, gp, bep, (float)N, HIDC, scale, shift);
  bn_apply_relu<<<rowVec, T, 0, stream>>>(bufA, scale, shift, bufB, N);

  gemm_wmma<PROJC, HIDC><<<gBlocks, T, 0, stream>>>(bufB, Wtp2, bp2, out, Mtiles);
  l2norm<<<cdiv(N, 8), T, 0, stream>>>(out, N);
}
